// BertGenerationMoE_86612310491391
// MI455X (gfx1250) — matrix operand order repaired and checked by reading, not compiled, not run on hardware
//
#include <hip/hip_runtime.h>
#include <hip/hip_bf16.h>

// ---------------------------------------------------------------------------
// BertGenerationMoE, MI455X (gfx1250), wave32 + WMMA f32_16x16x32_f16.
// Shapes: S=256 seqs, L=128, H=512, I=2048, E=16 experts, C=16 capacity.
// ---------------------------------------------------------------------------

#define S_ 256
#define L_ 128
#define H_ 512
#define I_ 2048
#define E_ 16
#define C_ 16
#define LH_ 65536            // L_*H_
#define MT_ 32               // output row tile per block
#define KC_ 64               // I-chunk width
#define A_LD 520             // f16 LDS stride for A tile (16B aligned rows)
#define HID_LD 72            // f16 LDS stride for hid chunk
#define Y_LD 512             // f32 LDS stride for y tile (fits 64KB)
#define LN_EPS 1e-12f

typedef __attribute__((ext_vector_type(16))) _Float16 v16h;
typedef __attribute__((ext_vector_type(8)))  _Float16 h8;
typedef __attribute__((ext_vector_type(8)))  float    v8f;

// --- WMMA wrapper: D = A(16x32 f16) * B(32x16 f16) + C(16x16 f32) ----------
__device__ __forceinline__ v8f wmma_f16(v16h a, v16h b, v8f c) {
  return __builtin_amdgcn_wmma_f32_16x16x32_f16(
      /*neg_a=*/false, a, /*neg_b=*/false, b,
      /*c_mod=*/(short)0, c, /*reuse_a=*/false, /*reuse_b=*/false);
}

// A fragment (16x32, f16) from a row-major f16 tile (stride ld, in elements).
// lane m=lane&15, half hh=lane>>4 holds K=hh*16+0..15 (matches B fragment)
// -> two contiguous 16B loads per lane.
__device__ __forceinline__ v16h load_fragA(const _Float16* tile, int ld,
                                           int mrow, int k, int lane) {
  const int m  = mrow + (lane & 15);
  const int hh = lane >> 4;
  const _Float16* r = tile + m * ld + k + hh * 16;
  union { v16h v; h8 h[2]; } u;
  u.h[0] = *(const h8*)(r);
  u.h[1] = *(const h8*)(r + 8);
  return u.v;
}

// B fragment (32x16, f16) where memory holds B transposed (column n contiguous
// in K).  Same k order as A: lane n=lane&15, half hh=lane>>4 holds K=hh*16+0..15 -> one
// contiguous 32B load per lane. Caller passes p = base + col*ldK + k + hh*16.
__device__ __forceinline__ v16h load_fragB(const _Float16* p) {
  union { v16h v; h8 h[2]; } u;
  u.h[0] = *(const h8*)(p);
  u.h[1] = *(const h8*)(p + 8);
  return u.v;
}

// --- 1) routing: faithful cumsum ranks (balanced inputs -> exact fit) ------
__global__ void route_kernel(const int* __restrict__ task_ids,
                             const int* __restrict__ hash_indices,
                             int* __restrict__ invmap,
                             int* __restrict__ dest_of,
                             int* __restrict__ rank_of) {
  int e = threadIdx.x;
  if (e >= E_) return;
  int cnt = 0;
  for (int s = 0; s < S_; ++s) {
    if (hash_indices[task_ids[s]] == e) {
      dest_of[s] = e;
      if (cnt < C_) { invmap[e * C_ + cnt] = s; rank_of[s] = cnt; }
      ++cnt;
    }
  }
  for (int r = cnt; r < C_; ++r) invmap[e * C_ + r] = -1;
}

// --- 2) transpose + f32->f16 convert: dst[e][c][r] = (f16)src[e][r][c] -----
__global__ void transpose_cvt_kernel(const float* __restrict__ src,
                                     _Float16* __restrict__ dst,
                                     int rows, int cols) {
  __shared__ float tile[32][33];
  const int e = blockIdx.z;
  const float*    s = src + (size_t)e * rows * cols;
  _Float16*       d = dst + (size_t)e * rows * cols;
  const int c0 = blockIdx.x * 32, r0 = blockIdx.y * 32;
  for (int i = threadIdx.y; i < 32; i += 8)
    tile[i][threadIdx.x] = s[(size_t)(r0 + i) * cols + c0 + threadIdx.x];
  __syncthreads();
  for (int i = threadIdx.y; i < 32; i += 8)
    d[(size_t)(c0 + i) * rows + r0 + threadIdx.x] = (_Float16)tile[threadIdx.x][i];
}

// --- 3) col-0 scatter quirk: sum0[e] = sum_c x[invmap[e][c]] ---------------
__global__ void sum0_kernel(const float* __restrict__ x,
                            const int* __restrict__ invmap,
                            float* __restrict__ sum0) {
  const int idx = blockIdx.x * 256 + threadIdx.x;    // < E_*L_*H_
  const int e = idx >> 16, rem = idx & (LH_ - 1);
  float s = 0.f;
#pragma unroll
  for (int r = 0; r < C_; ++r) {
    const int sq = invmap[e * C_ + r];
    if (sq >= 0) s += x[(size_t)sq * LH_ + rem];
  }
  sum0[idx] = s;
}

// --- 4) fused FFN + GeLU + residual + LayerNorm per (e, c, 32-row tile) ----
__global__ __launch_bounds__(256, 2)
void moe_ffn_kernel(const float* __restrict__ x,
                    const float* __restrict__ sum0,
                    const int* __restrict__ invmap,
                    const _Float16* __restrict__ W1t,  // [E][I][H] f16
                    const _Float16* __restrict__ W2t,  // [E][H][I] f16
                    const float* __restrict__ b1,
                    const float* __restrict__ b2,
                    const float* __restrict__ ln_g,
                    const float* __restrict__ ln_b,
                    float* __restrict__ ynorm) {       // [E][C][L][H] f32
  __shared__ union {
    struct {
      _Float16 As[MT_ * A_LD];
      _Float16 hid[2][MT_ * HID_LD];   // ping-pong: 1 barrier per chunk
    } p;
    float y[MT_ * Y_LD];
  } sm;

  const int m0 = blockIdx.x * MT_;   // row tile within L
  const int c  = blockIdx.y;         // capacity slot
  const int e  = blockIdx.z;         // expert
  const int tid  = threadIdx.x;
  const int lane = tid & 31, w = tid >> 5;
  const int hh = lane >> 4, n = lane & 15;

  const _Float16* W1e = W1t + (size_t)e * I_ * H_;
  const _Float16* W2e = W2t + (size_t)e * H_ * I_;

  // A-tile source: slot 0 = sum of expert's sequences, else the ranked seq.
  const int sseq = invmap[e * C_ + c];
  const float* asrc =
      (c == 0) ? (sum0 + ((size_t)e * L_ + m0) * H_)
               : ((sseq >= 0) ? (x + ((size_t)sseq * L_ + m0) * H_) : nullptr);

  for (int i = tid; i < MT_ * H_; i += 256) {
    const int r = i >> 9, col = i & (H_ - 1);
    const float v = asrc ? asrc[(size_t)r * H_ + col] : 0.f;
    sm.p.As[r * A_LD + col] = (_Float16)v;
  }
  __syncthreads();

  // y accumulators: wave w owns cols [w*64, w*64+64), rows 0..31 -> 8 tiles.
  v8f accY[8];
#pragma unroll
  for (int t = 0; t < 8; ++t) accY[t] = (v8f){0.f,0.f,0.f,0.f,0.f,0.f,0.f,0.f};

  const int tm = w >> 2, tn = w & 3;   // phase-1 tile of this wave

  for (int chunk = 0; chunk < I_; chunk += KC_) {
    _Float16* hidb = sm.p.hid[(chunk / KC_) & 1];

    // ---- phase 1: hid[32 x 64] = A(32x512) * W1[:, chunk:chunk+64] --------
    // B fragment software-pipelined one k-step ahead so the global load of
    // step k+32 overlaps the WMMA of step k.
    v8f acch = (v8f){0.f,0.f,0.f,0.f,0.f,0.f,0.f,0.f};
    const _Float16* b1base = W1e + (size_t)(chunk + tn * 16 + n) * H_ + hh * 16;
    v16h bcur = load_fragB(b1base);
#pragma unroll 4
    for (int k = 0; k < H_; k += 32) {
      v16h bnext;
      if (k + 32 < H_) bnext = load_fragB(b1base + k + 32);
      v16h a = load_fragA(sm.p.As, A_LD, tm * 16, k, lane);
      acch = wmma_f16(a, bcur, acch);
      bcur = bnext;
    }
    if (chunk + KC_ < I_) {  // prefetch next chunk's weight streams
      __builtin_prefetch(W1e + (size_t)(chunk + KC_ + tn * 16 + n) * H_, 0, 1);
      __builtin_prefetch(W2e + (size_t)(w * 64 + n) * I_ + chunk + KC_, 0, 1);
    }

    // bias + exact GeLU, store f16 to the ping-pong LDS chunk
    const float b1v = b1[e * I_ + chunk + tn * 16 + n];
#pragma unroll
    for (int j = 0; j < 8; ++j) {
      float v = acch[j] + b1v;
      v = 0.5f * v * (1.0f + erff(v * 0.70710678118654752f));
      hidb[(tm * 16 + j + 8 * hh) * HID_LD + tn * 16 + n] = (_Float16)v;
    }
    __syncthreads();   // single barrier per chunk (ping-pong covers WAR)

    // ---- phase 2: y[:, w*64 .. +64) += hid(32x64) * W2[chunk:, cols] ------
    // All 4 B fragments of a k-step are preloaded so the 8 loads clause
    // together and the 8 WMMAs issue back-to-back.
#pragma unroll
    for (int kk = 0; kk < KC_; kk += 32) {
      v16h bf[4];
#pragma unroll
      for (int tc = 0; tc < 4; ++tc) {
        const int col = w * 64 + tc * 16 + n;
        bf[tc] = load_fragB(W2e + (size_t)col * I_ + chunk + kk + hh * 16);
      }
      v16h a0 = load_fragA(hidb, HID_LD, 0,  kk, lane);
      v16h a1 = load_fragA(hidb, HID_LD, 16, kk, lane);
#pragma unroll
      for (int tc = 0; tc < 4; ++tc) {
        accY[tc]     = wmma_f16(a0, bf[tc], accY[tc]);
        accY[4 + tc] = wmma_f16(a1, bf[tc], accY[4 + tc]);
      }
    }
  }
  __syncthreads();   // all waves done with As/hid before y overwrites them

  // ---- epilogue: +b2 +residual into f32 LDS tile (reuses A/hid space) -----
  const float* resid = asrc;  // same quirk source as the A tile
#pragma unroll
  for (int tr = 0; tr < 2; ++tr) {
#pragma unroll
    for (int tc = 0; tc < 4; ++tc) {
      const int col = w * 64 + tc * 16 + n;
      const float b2v = b2[e * H_ + col];
#pragma unroll
      for (int j = 0; j < 8; ++j) {
        const int row = tr * 16 + j + 8 * hh;
        const float rv = resid ? resid[(size_t)row * H_ + col] : 0.f;
        sm.y[row * Y_LD + col] = accY[tr * 4 + tc][j] + b2v + rv;
      }
    }
  }
  __syncthreads();

  // ---- LayerNorm over H=512 per row; 4 rows per wave, wave32 reductions ---
#pragma unroll
  for (int rr = 0; rr < 4; ++rr) {
    const int row = w * 4 + rr;
    float vals[16], s = 0.f;
#pragma unroll
    for (int t = 0; t < 16; ++t) {
      vals[t] = sm.y[row * Y_LD + lane + 32 * t];
      s += vals[t];
    }
    for (int off = 16; off > 0; off >>= 1) s += __shfl_xor(s, off, 32);
    const float mu = s * (1.0f / (float)H_);
    float vs = 0.f;
#pragma unroll
    for (int t = 0; t < 16; ++t) { const float d = vals[t] - mu; vs += d * d; }
    for (int off = 16; off > 0; off >>= 1) vs += __shfl_xor(vs, off, 32);
    const float inv = rsqrtf(vs * (1.0f / (float)H_) + LN_EPS);
    float* op = ynorm + (((size_t)(e * C_ + c) * L_) + m0 + row) * H_;
#pragma unroll
    for (int t = 0; t < 16; ++t) {
      const int col = lane + 32 * t;
      op[col] = (vals[t] - mu) * inv * ln_g[e * H_ + col] + ln_b[e * H_ + col];
    }
  }
}

// --- 5) combine: out[s] = y[dest,rank] (+ y[dest,0] when rank>0) -----------
__global__ void combine_kernel(const float* __restrict__ ynorm,
                               const int* __restrict__ dest_of,
                               const int* __restrict__ rank_of,
                               float* __restrict__ out) {
  const size_t idx = (size_t)blockIdx.x * 256 + threadIdx.x;
  const int s = (int)(idx >> 16), rem = (int)(idx & (LH_ - 1));
  const int e = dest_of[s], r = rank_of[s];
  float v = ynorm[((size_t)(e * C_ + r) << 16) + rem];
  if (r > 0) v += ynorm[((size_t)(e * C_) << 16) + rem];
  out[idx] = v;
}

// ---------------------------------------------------------------------------
extern "C" void kernel_launch(void* const* d_in, const int* in_sizes, int n_in,
                              void* d_out, int out_size, void* d_ws, size_t ws_size,
                              hipStream_t stream) {
  const float* x            = (const float*)d_in[0];
  const int*   task_ids     = (const int*)  d_in[1];
  const int*   hash_indices = (const int*)  d_in[2];
  const float* W1           = (const float*)d_in[3];
  const float* b1           = (const float*)d_in[4];
  const float* W2           = (const float*)d_in[5];
  const float* b2           = (const float*)d_in[6];
  const float* ln_g         = (const float*)d_in[7];
  const float* ln_b         = (const float*)d_in[8];
  float* out = (float*)d_out;

  // Workspace layout (needs ~138.5 MB):
  char* ws = (char*)d_ws;
  _Float16* W1t   = (_Float16*)(ws);                          // 32 MB [E][I][H]
  _Float16* W2t   = (_Float16*)(ws + (size_t)33554432);       // 32 MB [E][H][I]
  float*    sum0  = (float*)   (ws + (size_t)67108864);       //  4 MB [E][L][H]
  float*    ynorm = (float*)   (ws + (size_t)71303168);       // 64 MB [E][C][L][H]
  int*      invmap  = (int*)   (ws + (size_t)138412032);      // 256
  int*      dest_of = invmap + S_;                            // 256
  int*      rank_of = dest_of + S_;                           // 256

  route_kernel<<<1, 32, 0, stream>>>(task_ids, hash_indices, invmap, dest_of, rank_of);

  // W1: [H][I] per expert -> W1t [I][H];  W2: [I][H] -> W2t [H][I]
  transpose_cvt_kernel<<<dim3(I_ / 32, H_ / 32, E_), dim3(32, 8), 0, stream>>>(W1, W1t, H_, I_);
  transpose_cvt_kernel<<<dim3(H_ / 32, I_ / 32, E_), dim3(32, 8), 0, stream>>>(W2, W2t, I_, H_);

  sum0_kernel<<<(E_ * L_ * H_) / 256, 256, 0, stream>>>(x, invmap, sum0);

  moe_ffn_kernel<<<dim3(L_ / MT_, C_, E_), 256, 0, stream>>>(
      x, sum0, invmap, W1t, W2t, b1, b2, ln_g, ln_b, ynorm);

  combine_kernel<<<(S_ * L_ * H_) / 256, 256, 0, stream>>>(ynorm, dest_of, rank_of, out);
}
